// GraphAttentionLayerV2_52484500357560
// MI455X (gfx1250) — matrix operand order repaired and checked by reading, not compiled, not run on hardware
//
#include <hip/hip_runtime.h>
#include <hip/hip_bf16.h>
#include <math.h>

// ---------------------------------------------------------------------------
// GATv2 multi-relation layer for MI455X (gfx1250, wave32, WMMA)
// B=2, N=768, IN_F=256, H=4, D=16, R=3, OUT_F=256
// ---------------------------------------------------------------------------

typedef __attribute__((ext_vector_type(16))) __bf16 v16bf;
typedef __attribute__((ext_vector_type(8)))  __bf16 v8bf;
typedef __attribute__((ext_vector_type(8)))  float  v8f;

#define Bsz   2
#define Nn    768
#define INF_  256
#define HD    64
#define Dd    16
#define Rr    3
#define BN    (Bsz*Nn)          // 1536
#define OUTF  256
#define SLOPE 0.01f
#define NEGV  (-9e15f)

#define IT    16                // destination rows (i) per block
#define JT    64                // source rows (j) per LDS tile
#define NJT   (Nn/JT)           // 12 tiles
#define LPAD  68                // row stride (floats): 272B = 16B aligned, bank step 4

// ------------------------------- prep kernels ------------------------------

__global__ void prep_v(const float* __restrict__ V, __bf16* __restrict__ Vbf) {
  int idx = blockIdx.x * 256 + threadIdx.x;            // 393216
  Vbf[idx] = (__bf16)V[idx];
}

// W_src/W_dst [3][256][64] fp32 -> Wt bf16 [6][64][256]  (col-major per matrix)
__global__ void prep_w(const float* __restrict__ Wsrc, const float* __restrict__ Wdst,
                       __bf16* __restrict__ Wt) {
  int idx = blockIdx.x * 256 + threadIdx.x;            // 6*64*256 = 98304
  int m   = idx >> 14;
  int rem = idx & 16383;
  int n   = rem >> 8;                                  // col 0..63
  int k   = rem & 255;                                 // K   0..255
  int r   = m >> 1;
  const float* W = (m & 1) ? Wdst : Wsrc;
  Wt[idx] = (__bf16)W[(size_t)r * (INF_*HD) + (size_t)k * HD + n];
}

// squeeze_W [48][256] fp32 -> SqT bf16 [256][64] (col-major, K padded 48->64)
__global__ void prep_sq(const float* __restrict__ sqW, __bf16* __restrict__ SqT) {
  int idx = blockIdx.x * 256 + threadIdx.x;            // 16384
  int n = idx >> 6;
  int k = idx & 63;
  SqT[idx] = (k < 48) ? (__bf16)sqW[(size_t)k * OUTF + n] : (__bf16)0.f;
}

// ------------------------- projection GEMMs (WMMA) -------------------------
// F[m][row][col] = sum_k Vbf[row][k] * W[m][k][col],  m in 0..5 (r*2 + src/dst)
__global__ void __launch_bounds__(256) proj_wmma(const __bf16* __restrict__ Vbf,
                                                 const __bf16* __restrict__ Wt,
                                                 float* __restrict__ F) {
  int wave = threadIdx.x >> 5, lane = threadIdx.x & 31;
  int t  = blockIdx.x * 8 + wave;                      // 0..2303
  int nt = t & 3;
  int mt = (t >> 2) % 96;
  int m  = (t >> 2) / 96;
  int lo = lane & 15, hi = lane >> 4;

  const __bf16* Ab = Vbf + ((size_t)(mt * 16 + lo)) * INF_ + hi * 16;
  const __bf16* Bb = Wt + (size_t)m * (HD * INF_) + ((size_t)(nt * 16 + lo)) * INF_ + hi * 16;

  v8f c = {0.f,0.f,0.f,0.f,0.f,0.f,0.f,0.f};
#pragma unroll
  for (int k = 0; k < 8; ++k) {
    v16bf a, b;
    *((v8bf*)&a)     = *(const v8bf*)(Ab + k * 32);
    *((v8bf*)&a + 1) = *(const v8bf*)(Ab + k * 32 + 8);
    b = *(const v16bf*)(Bb + k * 32);
    c = __builtin_amdgcn_wmma_f32_16x16x32_bf16(false, a, false, b, (short)0, c,
                                                false, false);
  }
  float* Dp = F + ((size_t)m * BN + mt * 16 + hi * 8) * HD + nt * 16 + lo;
#pragma unroll
  for (int i = 0; i < 8; ++i) Dp[(size_t)i * HD] = c[i];
}

// ---- fused scores / online softmax / aggregation / LayerNorm (tiled) ------
// one block per (b, r, 16-row i-tile); j streamed in 64-row LDS tiles shared
// by all 16 i's (16x less L2 traffic than one-i-per-block).
// thread t: il = t>>4 (i row), jg = t&15 (j lane); i-group == shfl width-16 seg.
__global__ void __launch_bounds__(256) gat_attn_tiled(
    const float* __restrict__ F, const float* __restrict__ adj,
    const float* __restrict__ Arel, const float* __restrict__ gamma,
    const float* __restrict__ beta, float* __restrict__ OutCat) {
  int itile = blockIdx.x % (Nn / IT);
  int r = (blockIdx.x / (Nn / IT)) % Rr;
  int b = blockIdx.x / ((Nn / IT) * Rr);
  int tid = threadIdx.x;
  int il = tid >> 4;          // 0..15
  int jg = tid & 15;          // 0..15

  __shared__ float s_fsi[IT][LPAD];
  __shared__ float s_fd[JT][LPAD];
  __shared__ float s_fs[JT][LPAD];
  __shared__ float s_A[HD];

  const float* FS = F + ((size_t)(r * 2)     * BN + b * Nn) * HD;
  const float* FD = F + ((size_t)(r * 2 + 1) * BN + b * Nn) * HD;
  int i0 = itile * IT;
  int i_glob = i0 + il;

  if (tid < HD) s_A[tid] = Arel[r * HD + tid];
  {
    int row = tid >> 4, c = (tid & 15) * 4;   // 16 rows x 16 floats4
    *(float4*)&s_fsi[row][c] = *(const float4*)(FS + (size_t)(i0 + row) * HD + c);
  }

  const float* adj_row = adj + (((size_t)b * Nn + i_glob) * 2 + r) * Nn;  // r<2 only

  float m[4], l[4], acc[4][Dd];
#pragma unroll
  for (int h = 0; h < 4; ++h) { m[h] = -INFINITY; l[h] = 0.f; }
#pragma unroll
  for (int h = 0; h < 4; ++h)
#pragma unroll
    for (int d = 0; d < Dd; ++d) acc[h][d] = 0.f;

  int srow = tid >> 2;              // 0..63 (staging row)
  int sseg = (tid & 3) * 16;        // 0,16,32,48 (staging 16-float chunk)

  for (int jt = 0; jt < NJT; ++jt) {
    int jbase = jt * JT;
    __syncthreads();                // previous tile fully consumed
    {
      const float* fdsrc = FD + (size_t)(jbase + srow) * HD + sseg;
      const float* fssrc = FS + (size_t)(jbase + srow) * HD + sseg;
#pragma unroll
      for (int c = 0; c < 16; c += 4) {
        *(float4*)&s_fd[srow][sseg + c] = *(const float4*)(fdsrc + c);
        *(float4*)&s_fs[srow][sseg + c] = *(const float4*)(fssrc + c);
      }
      if (jt + 1 < NJT) {           // gfx1250 global_prefetch_b8 of next tile
        __builtin_prefetch(fdsrc + (size_t)JT * HD, 0, 1);
        __builtin_prefetch(fssrc + (size_t)JT * HD, 0, 1);
      }
    }
    __syncthreads();

    // ---- scores for this thread's 4 j's (jl = jg + 16q)
    float e[4][4];                  // [q][h]
    float tmax[4] = {-INFINITY, -INFINITY, -INFINITY, -INFINITY};
#pragma unroll
    for (int q = 0; q < 4; ++q) {
      int jl = jg + 16 * q;
      int j  = jbase + jl;
      bool valid = (r < 2) ? (adj_row[j] > 0.f) : (j == i_glob);
#pragma unroll
      for (int h = 0; h < 4; ++h) {
        float s = 0.f;
#pragma unroll
        for (int d = 0; d < Dd; ++d) {
          float x = s_fsi[il][h * Dd + d] + s_fd[jl][h * Dd + d];
          x = (x > 0.f) ? x : SLOPE * x;
          s += s_A[h * Dd + d] * x;
        }
        e[q][h] = valid ? s : NEGV;
        tmax[h] = fmaxf(tmax[h], e[q][h]);
      }
    }
    // group max over the 16 threads sharing this i (width-16 butterfly)
#pragma unroll
    for (int off = 8; off; off >>= 1)
#pragma unroll
      for (int h = 0; h < 4; ++h)
        tmax[h] = fmaxf(tmax[h], __shfl_xor(tmax[h], off, 16));

    // ---- online softmax update
    float corr[4], tsum[4];
#pragma unroll
    for (int h = 0; h < 4; ++h) {
      float mn = fmaxf(m[h], tmax[h]);
      corr[h] = __expf(m[h] - mn);  // first tile: exp(-inf)=0
      m[h] = mn;
      tsum[h] = 0.f;
    }
#pragma unroll
    for (int q = 0; q < 4; ++q)
#pragma unroll
      for (int h = 0; h < 4; ++h) {
        e[q][h] = __expf(e[q][h] - m[h]);
        tsum[h] += e[q][h];
      }
#pragma unroll
    for (int off = 8; off; off >>= 1)
#pragma unroll
      for (int h = 0; h < 4; ++h) tsum[h] += __shfl_xor(tsum[h], off, 16);
#pragma unroll
    for (int h = 0; h < 4; ++h) l[h] = l[h] * corr[h] + tsum[h];

    // ---- rescale + accumulate per-head
#pragma unroll
    for (int h = 0; h < 4; ++h)
#pragma unroll
      for (int d = 0; d < Dd; ++d) acc[h][d] *= corr[h];
#pragma unroll
    for (int q = 0; q < 4; ++q) {
      int jl = jg + 16 * q;
#pragma unroll
      for (int h = 0; h < 4; ++h) {
        float eq = e[q][h];
#pragma unroll
        for (int d = 0; d < Dd; ++d) acc[h][d] += eq * s_fs[jl][h * Dd + d];
      }
    }
  }

  // ---- fold heads (divide by softmax denom), reduce over the 16-thread group
  float tmp[Dd];
  float rl0 = 1.f / l[0], rl1 = 1.f / l[1], rl2 = 1.f / l[2], rl3 = 1.f / l[3];
#pragma unroll
  for (int d = 0; d < Dd; ++d)
    tmp[d] = acc[0][d] * rl0 + acc[1][d] * rl1 + acc[2][d] * rl2 + acc[3][d] * rl3;
#pragma unroll
  for (int off = 8; off; off >>= 1)
#pragma unroll
    for (int d = 0; d < Dd; ++d) tmp[d] += __shfl_xor(tmp[d], off, 16);

  if (jg == 0) {  // one thread per i: LayerNorm(16) + LeakyReLU + store
    float mu = 0.f;
#pragma unroll
    for (int d = 0; d < Dd; ++d) mu += tmp[d];
    mu *= (1.f / Dd);
    float var = 0.f;
#pragma unroll
    for (int d = 0; d < Dd; ++d) { float dx = tmp[d] - mu; var += dx * dx; }
    var *= (1.f / Dd);
    float rs = rsqrtf(var + 1e-5f);
    float* op = OutCat + ((size_t)b * Nn + i_glob) * (Dd * Rr) + r * Dd;
#pragma unroll
    for (int d = 0; d < Dd; ++d) {
      float y = (tmp[d] - mu) * rs * gamma[r * Dd + d] + beta[r * Dd + d];
      y = (y > 0.f) ? y : SLOPE * y;
      op[d] = y;
    }
  }
}

// ----------------- convert concat output -> bf16 (K pad 48->64) ------------
__global__ void cvt_out(const float* __restrict__ OutCat, __bf16* __restrict__ OutBf) {
  int idx = blockIdx.x * 256 + threadIdx.x;            // 1536*64 = 98304
  int row = idx >> 6, k = idx & 63;
  OutBf[idx] = (k < 48) ? (__bf16)OutCat[(size_t)row * 48 + k] : (__bf16)0.f;
}

// ------------- squeeze GEMM (WMMA) + bias + residual, fused store ----------
__global__ void __launch_bounds__(256) squeeze_wmma(const __bf16* __restrict__ OutBf,
                                                    const __bf16* __restrict__ SqT,
                                                    const float* __restrict__ sqb,
                                                    const float* __restrict__ V,
                                                    float* __restrict__ out) {
  int wave = threadIdx.x >> 5, lane = threadIdx.x & 31;
  int t  = blockIdx.x * 8 + wave;                      // 0..1535
  int nt = t & 15;
  int mt = t >> 4;
  int lo = lane & 15, hi = lane >> 4;

  const __bf16* Ab = OutBf + ((size_t)(mt * 16 + lo)) * 64 + hi * 16;
  const __bf16* Bb = SqT   + ((size_t)(nt * 16 + lo)) * 64 + hi * 16;

  v8f c = {0.f,0.f,0.f,0.f,0.f,0.f,0.f,0.f};
#pragma unroll
  for (int k = 0; k < 2; ++k) {
    v16bf a, b;
    *((v8bf*)&a)     = *(const v8bf*)(Ab + k * 32);
    *((v8bf*)&a + 1) = *(const v8bf*)(Ab + k * 32 + 8);
    b = *(const v16bf*)(Bb + k * 32);
    c = __builtin_amdgcn_wmma_f32_16x16x32_bf16(false, a, false, b, (short)0, c,
                                                false, false);
  }
  int row = mt * 16 + hi * 8;
  int col = nt * 16 + lo;
  float bias = sqb[col];
  const float* vp = V   + (size_t)row * OUTF + col;
  float*       op = out + (size_t)row * OUTF + col;
#pragma unroll
  for (int i = 0; i < 8; ++i) op[(size_t)i * OUTF] = c[i] + bias + vp[(size_t)i * OUTF];
}

// ---------------------------------------------------------------------------

extern "C" void kernel_launch(void* const* d_in, const int* in_sizes, int n_in,
                              void* d_out, int out_size, void* d_ws, size_t ws_size,
                              hipStream_t stream) {
  (void)in_sizes; (void)n_in; (void)out_size; (void)ws_size;
  const float* V     = (const float*)d_in[0];
  const float* adj   = (const float*)d_in[1];
  const float* W_src = (const float*)d_in[2];
  const float* W_dst = (const float*)d_in[3];
  const float* Arel  = (const float*)d_in[4];
  const float* gamma = (const float*)d_in[5];
  const float* beta  = (const float*)d_in[6];
  const float* sqW   = (const float*)d_in[7];
  const float* sqb   = (const float*)d_in[8];
  float* out = (float*)d_out;
  char* ws = (char*)d_ws;

  // workspace layout (bytes)
  __bf16* Vbf    = (__bf16*)(ws + 0);          // 1536*256*2      = 786432
  __bf16* Wt     = (__bf16*)(ws + 786432);     // 6*64*256*2      = 196608
  float*  F      = (float* )(ws + 983040);     // 6*1536*64*4     = 2359296
  float*  OutCat = (float* )(ws + 3342336);    // 1536*48*4       = 294912
  __bf16* OutBf  = (__bf16*)(ws + 3637248);    // 1536*64*2       = 196608
  __bf16* SqT    = (__bf16*)(ws + 3833856);    // 256*64*2        = 32768

  prep_v  <<<1536, 256, 0, stream>>>(V, Vbf);
  prep_w  <<< 384, 256, 0, stream>>>(W_src, W_dst, Wt);
  prep_sq <<<  64, 256, 0, stream>>>(sqW, SqT);

  proj_wmma<<<288, 256, 0, stream>>>(Vbf, Wt, F);                     // 2304 tiles / 8 waves
  gat_attn_tiled<<<Bsz * Rr * (Nn / IT), 256, 0, stream>>>(F, adj, Arel, gamma,
                                                           beta, OutCat);  // 288 blocks
  cvt_out  <<<384, 256, 0, stream>>>(OutCat, OutBf);
  squeeze_wmma<<<192, 256, 0, stream>>>(OutBf, SqT, sqb, V, out);     // 1536 tiles / 8 waves
}
